// CausalSelfAttention_65524021067825
// MI455X (gfx1250) — matrix operand order repaired and checked by reading, not compiled, not run on hardware
//
#include <hip/hip_runtime.h>
#include <math.h>
#include <stdint.h>

typedef __bf16 bf16;
typedef __attribute__((ext_vector_type(16))) __bf16 v16bf;
typedef __attribute__((ext_vector_type(8)))  float  v8f;

union AF {
  v16bf v;
  uint4 q[2];
};

__device__ inline v8f wmma_bf16(const AF& a, const AF& b, v8f c) {
  return __builtin_amdgcn_wmma_f32_16x16x32_bf16(false, a.v, false, b.v,
                                                 (short)0, c, false, false);
}

// ---------------- conversion kernels ----------------

__global__ __launch_bounds__(256)
void cvt_f32_to_bf16(const float* __restrict__ in, bf16* __restrict__ out, int n) {
  int i = blockIdx.x * 256 + threadIdx.x;
  if (i < n) out[i] = (bf16)in[i];
}

// in: [K][N] f32 row-major  ->  out: [N][K] bf16 row-major (N x K)
__global__ __launch_bounds__(256)
void transpose_cvt(const float* __restrict__ in, bf16* __restrict__ out, int K, int N) {
  int i = blockIdx.x * 256 + threadIdx.x;
  if (i >= K * N) return;
  int n = i / K;
  int k = i - n * K;
  out[i] = (bf16)in[(size_t)k * N + n];
}

// ---------------- WMMA GEMM: C[M,N] = A[M,K] * Bt[N,K]^T + bias ----------------
// Block: 256 threads (8 waves), tile 128x128, K-slab 32, double-buffered LDS
// filled with CDNA5 async global->LDS copies (ASYNCcnt).
// Wave (wm in 0..1, wn in 0..3) computes 64 rows x 32 cols = 4x2 16x16 frags.

#define SLAB (128 * 40)   // one LDS slab in bf16 elements (40-elem padded rows)

__global__ __launch_bounds__(256)
void gemm_bf16_wmma(const bf16* __restrict__ A, const bf16* __restrict__ Bt,
                    const float* __restrict__ bias, void* __restrict__ out,
                    int M, int N, int K, int out_is_f32) {
  __shared__ __align__(16) bf16 As[2 * SLAB];
  __shared__ __align__(16) bf16 Bs[2 * SLAB];

  const int tid  = threadIdx.x;
  const int lane = tid & 31;
  const int w    = tid >> 5;
  const int wm   = w & 1;
  const int wn   = w >> 1;
  const int rowbase = blockIdx.y * 128;
  const int colbase = blockIdx.x * 128;
  const int lm = lane & 15, lg = lane >> 4;

  v8f acc[4][2];
  for (int a = 0; a < 4; ++a)
    for (int b = 0; b < 2; ++b)
      for (int e = 0; e < 8; ++e) acc[a][b][e] = 0.f;

  const int r  = tid >> 1;
  const int ch = (tid & 1) * 16;

  // per-thread source rows / LDS destinations for the async tile fill
  const bf16* gA = A  + (size_t)(rowbase + r) * K + ch;
  const bf16* gB = Bt + (size_t)(colbase + r) * K + ch;
  const uint32_t ldsA0 = (uint32_t)(uintptr_t)(As + r * 40 + ch);
  const uint32_t ldsB0 = (uint32_t)(uintptr_t)(Bs + r * 40 + ch);

  // issue one tile fill (4 async b128 per lane => wave ASYNCcnt += 4)
  auto issue = [&](int k0, int sel) {
    unsigned long long ga = (unsigned long long)(uintptr_t)(gA + k0);
    unsigned long long gb = (unsigned long long)(uintptr_t)(gB + k0);
    uint32_t la = ldsA0 + (uint32_t)(sel * SLAB * 2);
    uint32_t lb = ldsB0 + (uint32_t)(sel * SLAB * 2);
    asm volatile(
        "global_load_async_to_lds_b128 %0, %1, off\n\t"
        "global_load_async_to_lds_b128 %0, %1, off offset:16\n\t"
        "global_load_async_to_lds_b128 %2, %3, off\n\t"
        "global_load_async_to_lds_b128 %2, %3, off offset:16"
        :
        : "v"(la), "v"(ga), "v"(lb), "v"(gb)
        : "memory");
  };

  issue(0, 0);  // prologue: tile 0 into buffer 0

  for (int k0 = 0; k0 < K; k0 += 32) {
    const int cur = (k0 >> 5) & 1;
    if (k0 + 32 < K) {
      issue(k0 + 32, cur ^ 1);                       // prefetch next tile
      asm volatile("s_wait_asynccnt 0x4" ::: "memory");  // current tile landed
    } else {
      asm volatile("s_wait_asynccnt 0x0" ::: "memory");
    }
    __syncthreads();  // all waves' fills visible

    const bf16* Ac = As + cur * SLAB;
    const bf16* Bc = Bs + cur * SLAB;

    AF af[4], bfr[2];
    for (int mf = 0; mf < 4; ++mf) {
      const bf16* p = Ac + (wm * 64 + mf * 16 + lm) * 40;
      af[mf].q[0] = *(const uint4*)(p + 8 * lg);        // K = 8g .. 8g+7
      af[mf].q[1] = *(const uint4*)(p + 16 + 8 * lg);   // K = 16+8g .. +7
    }
    for (int nf = 0; nf < 2; ++nf) {
      const bf16* p = Bc + (wn * 32 + nf * 16 + lm) * 40;
      bfr[nf].q[0] = *(const uint4*)(p + 8 * lg);       // K = 8g .. 8g+7
      bfr[nf].q[1] = *(const uint4*)(p + 16 + 8 * lg);  // K = 16+8g .. +7
    }
    for (int mf = 0; mf < 4; ++mf)
      for (int nf = 0; nf < 2; ++nf)
        acc[mf][nf] = wmma_bf16(af[mf], bfr[nf], acc[mf][nf]);
    __syncthreads();  // WAR: buffer reused two tiles later
  }

  // epilogue: C lane holds col n = lm, rows i + 8*lg
  for (int nf = 0; nf < 2; ++nf) {
    int c = colbase + wn * 32 + nf * 16 + lm;
    float bv = bias ? bias[c] : 0.f;
    for (int mf = 0; mf < 4; ++mf) {
      for (int i = 0; i < 8; ++i) {
        int rr = rowbase + wm * 64 + mf * 16 + i + 8 * lg;
        float vv = acc[mf][nf][i] + bv;
        if (out_is_f32) ((float*)out)[(size_t)rr * N + c] = vv;
        else            ((bf16*)out)[(size_t)rr * N + c] = (bf16)vv;
      }
    }
  }
}

// ---------------- RoPE + head layout ----------------
// qkv: [B*T, 3072] bf16. Writes Q,K: [B,H,T,64] bf16 (rope applied),
// V transposed: [B,H,64,T] bf16.

__global__ __launch_bounds__(256)
void rope_split(const bf16* __restrict__ qkv, bf16* __restrict__ Q,
                bf16* __restrict__ Kk, bf16* __restrict__ VT) {
  int idx = blockIdx.x * 256 + threadIdx.x;
  if (idx >= 2 * 2048 * 16 * 32) return;
  int d = idx & 31;
  int h = (idx >> 5) & 15;
  int t = (idx >> 9) & 2047;
  int b = idx >> 20;

  const bf16* row = qkv + (size_t)(b * 2048 + t) * 3072;
  int cq = h * 64 + d;
  float q1 = (float)row[cq],        q2 = (float)row[cq + 32];
  float k1 = (float)row[1024 + cq], k2 = (float)row[1024 + cq + 32];

  // inv_freq = 10000^(-d/32) = exp(-d * ln(10000)/32)
  float inv = __expf(-(float)d * 0.28782313662425572f);
  float ang = (float)t * inv;
  float s, c;
  __sincosf(ang, &s, &c);

  size_t hb = ((size_t)(b * 16 + h)) * 2048 * 64 + (size_t)t * 64;
  Q[hb + d]       = (bf16)(q1 * c - q2 * s);
  Q[hb + d + 32]  = (bf16)(q1 * s + q2 * c);
  Kk[hb + d]      = (bf16)(k1 * c - k2 * s);
  Kk[hb + d + 32] = (bf16)(k1 * s + k2 * c);

  size_t vb = ((size_t)(b * 16 + h)) * 64 * 2048;
  VT[vb + (size_t)d * 2048 + t]        = row[2048 + cq];
  VT[vb + (size_t)(d + 32) * 2048 + t] = row[2048 + cq + 32];
}

// ---------------- flash attention (one wave per 16-row q tile) ----------------

__global__ __launch_bounds__(256)
void flash_attn(const bf16* __restrict__ Q, const bf16* __restrict__ K,
                const bf16* __restrict__ VT, bf16* __restrict__ Y) {
  __shared__ __align__(16) bf16 Pt[8 * 16 * 40];  // per-wave 16x32 P tile (stride 40)

  const int w    = threadIdx.x >> 5;
  const int lane = threadIdx.x & 31;
  const int qt   = blockIdx.x * 8 + w;          // 0 .. 4095
  const int qblk = qt & 127;
  const int h    = (qt >> 7) & 15;
  const int b    = qt >> 11;
  const int qbase = qblk * 16;
  const int lm = lane & 15, lg = lane >> 4;

  const bf16* Qh = Q  + ((size_t)(b * 16 + h)) * 2048 * 64;
  const bf16* Kh = K  + ((size_t)(b * 16 + h)) * 2048 * 64;
  const bf16* Vh = VT + ((size_t)(b * 16 + h)) * 64 * 2048;
  bf16* P = Pt + w * (16 * 40);

  // Q fragments (A operand): rows qbase..qbase+15, K-dim = hd (two 32-chunks)
  AF qf[2];
  {
    const bf16* qp = Qh + (size_t)(qbase + lm) * 64;
    qf[0].q[0] = *(const uint4*)(qp + 8 * lg);
    qf[0].q[1] = *(const uint4*)(qp + 16 + 8 * lg);
    qf[1].q[0] = *(const uint4*)(qp + 32 + 8 * lg);
    qf[1].q[1] = *(const uint4*)(qp + 48 + 8 * lg);
  }

  v8f o[4];
  for (int d = 0; d < 4; ++d) for (int e = 0; e < 8; ++e) o[d][e] = 0.f;
  float rm[8], rl[8];
  for (int i = 0; i < 8; ++i) { rm[i] = -__builtin_inff(); rl[i] = 0.f; }

  for (int kb = 0; kb < qbase + 16; kb += 32) {
    // S = Q * K^T for 32 keys (two 16-key B blocks), K rows act as N x K layout
    v8f s[2];
    for (int j = 0; j < 2; ++j) {
      AF kf0, kf1;
      const bf16* kp = Kh + (size_t)(kb + j * 16 + lm) * 64;
      kf0.q[0] = *(const uint4*)(kp + 8 * lg);
      kf0.q[1] = *(const uint4*)(kp + 16 + 8 * lg);
      kf1.q[0] = *(const uint4*)(kp + 32 + 8 * lg);
      kf1.q[1] = *(const uint4*)(kp + 48 + 8 * lg);
      v8f z;
      for (int e = 0; e < 8; ++e) z[e] = 0.f;
      z = wmma_bf16(qf[0], kf0, z);
      s[j] = wmma_bf16(qf[1], kf1, z);
    }

    // scale + causal mask + online softmax stats
    float nm[8], al[8];
    for (int i = 0; i < 8; ++i) {
      int rrow = qbase + i + 8 * lg;
      float a0 = s[0][i] * 0.125f;
      float a1 = s[1][i] * 0.125f;
      if (kb + lm > rrow)      a0 = -__builtin_inff();
      if (kb + 16 + lm > rrow) a1 = -__builtin_inff();
      s[0][i] = a0; s[1][i] = a1;
      float mx = fmaxf(a0, a1);
      for (int off = 1; off < 16; off <<= 1)
        mx = fmaxf(mx, __shfl_xor(mx, off, 32));
      nm[i] = fmaxf(rm[i], mx);
      al[i] = __expf(rm[i] - nm[i]);
      rm[i] = nm[i];
    }
    for (int j = 0; j < 2; ++j)
      for (int i = 0; i < 8; ++i)
        s[j][i] = __expf(s[j][i] - nm[i]);
    for (int i = 0; i < 8; ++i) {
      float sum = s[0][i] + s[1][i];
      for (int off = 1; off < 16; off <<= 1)
        sum += __shfl_xor(sum, off, 32);
      rl[i] = rl[i] * al[i] + sum;
    }

    // C-layout -> A-layout transpose of P through per-wave LDS tile
    for (int j = 0; j < 2; ++j)
      for (int i = 0; i < 8; ++i)
        P[(i + 8 * lg) * 40 + j * 16 + lm] = (bf16)s[j][i];

    for (int d = 0; d < 4; ++d)
      for (int i = 0; i < 8; ++i) o[d][i] *= al[i];

    AF pf;
    {
      const bf16* pp = P + lm * 40;
      pf.q[0] = *(const uint4*)(pp + 8 * lg);
      pf.q[1] = *(const uint4*)(pp + 16 + 8 * lg);
    }
    // O += P * V  (V^T rows are the B fragments, contiguous along keys)
    for (int dj = 0; dj < 4; ++dj) {
      AF vf;
      const bf16* vp = Vh + (size_t)(dj * 16 + lm) * 2048 + kb;
      vf.q[0] = *(const uint4*)(vp + 8 * lg);
      vf.q[1] = *(const uint4*)(vp + 16 + 8 * lg);
      o[dj] = wmma_bf16(pf, vf, o[dj]);
    }
  }

  // normalize and store to y[B*T, C] bf16
  for (int dj = 0; dj < 4; ++dj) {
    int c = h * 64 + dj * 16 + lm;
    for (int i = 0; i < 8; ++i) {
      int tq = qbase + i + 8 * lg;
      float vv = o[dj][i] / rl[i];
      Y[((size_t)(b * 2048 + tq)) * 1024 + c] = (bf16)vv;
    }
  }
}

// ---------------- host launch ----------------

extern "C" void kernel_launch(void* const* d_in, const int* in_sizes, int n_in,
                              void* d_out, int out_size, void* d_ws, size_t ws_size,
                              hipStream_t stream) {
  (void)in_sizes; (void)n_in; (void)out_size; (void)ws_size;
  const float* x      = (const float*)d_in[0];  // [2,2048,1024]
  const float* w_attn = (const float*)d_in[1];  // [1024,3072]
  const float* b_attn = (const float*)d_in[2];  // [3072]
  const float* w_proj = (const float*)d_in[3];  // [1024,1024]
  const float* b_proj = (const float*)d_in[4];  // [1024]
  float* out = (float*)d_out;                   // [4096,1024]

  const size_t NTOK = 4096, C = 1024, N3 = 3072;

  bf16* xb   = (bf16*)d_ws;             // NTOK*C
  bf16* watT = xb   + NTOK * C;         // N3*C   (N x K)
  bf16* wpT  = watT + N3 * C;           // C*C    (N x K)
  bf16* qkv  = wpT  + C * C;            // NTOK*N3
  bf16* Qb   = qkv  + NTOK * N3;        // NTOK*C  [B,H,T,64]
  bf16* Kb   = Qb   + NTOK * C;
  bf16* Vt   = Kb   + NTOK * C;         // [B,H,64,T]
  bf16* Yb   = Vt   + NTOK * C;         // NTOK*C

  // 1) precision conversion / weight transposes
  cvt_f32_to_bf16<<<(int)(NTOK * C / 256), 256, 0, stream>>>(x, xb, (int)(NTOK * C));
  transpose_cvt<<<(int)(C * N3 / 256), 256, 0, stream>>>(w_attn, watT, (int)C, (int)N3);
  transpose_cvt<<<(int)(C * C / 256), 256, 0, stream>>>(w_proj, wpT, (int)C, (int)C);

  // 2) qkv = x @ w_attn + b_attn  (bf16 out)
  gemm_bf16_wmma<<<dim3(24, 32), 256, 0, stream>>>(xb, watT, b_attn, qkv,
                                                   (int)NTOK, (int)N3, (int)C, 0);

  // 3) RoPE + head-major layout (V transposed)
  rope_split<<<(2 * 2048 * 16 * 32) / 256, 256, 0, stream>>>(qkv, Qb, Kb, Vt);

  // 4) causal flash attention
  flash_attn<<<512, 256, 0, stream>>>(Qb, Kb, Vt, Yb);

  // 5) out = y @ w_proj + b_proj  (f32 out)
  gemm_bf16_wmma<<<dim3(8, 32), 256, 0, stream>>>(Yb, wpT, b_proj, out,
                                                  (int)NTOK, (int)C, (int)C, 1);
}
